// AbsorberPathAggregator_56564719289129
// MI455X (gfx1250) — matrix operand order repaired and checked by reading, not compiled, not run on hardware
//
#include <hip/hip_runtime.h>
#include <hip/hip_bf16.h>
#include <math.h>

// ---------------------------------------------------------------------------
// AbsorberPathAggregator for MI455X (gfx1250, wave32, WMMA 16x16x32 f16)
//
// Dominant cost: element-pair MLP (B*P*nE = 524288 rows, 96->128->128->64,
// ~38.7 GFLOP). Fused with the P-reduction so g_elem (134 MB) never touches
// HBM. All dense layers run on v_wmma_f32_16x16x32_f16 (f32 accumulate).
// Round 2: weight B-fragments + biases are hoisted into registers in the
// fused kernel (block-invariant), so the inner loop is ds_load + v_wmma only.
// ---------------------------------------------------------------------------

#define BB 16
#define NN 64
#define HH 128
#define PP 256
#define NE 128
#define NPAIR 2016   // 64*63/2
#define KGEOM 384    // 353 padded to mult of 32
#define CUT 6.0f

typedef _Float16 v16h __attribute__((ext_vector_type(16)));
typedef _Float16 v8h  __attribute__((ext_vector_type(8)));
typedef float    v8f  __attribute__((ext_vector_type(8)));

__device__ __forceinline__ float silu_f(float x) { return x / (1.0f + __expf(-x)); }
__device__ __forceinline__ float cutoff_f(float r) {
  return (r <= CUT) ? 0.5f * (__cosf(3.14159265358979323846f * r / CUT) + 1.0f) : 0.0f;
}

// --- WMMA fragment loaders (contiguous k order, same for A and B, wave32) -------------------
// A (16x32, MxK): lane<16 -> K kk..kk+15, lane>=16 -> K kk+16..kk+31, contiguous per lane
__device__ __forceinline__ v16h load_a_frag(const _Float16* A, int ldA, int mr,
                                            int kk, int lane) {
  int m  = mr + (lane & 15);
  int kb = (lane < 16) ? 0 : 16;
  const _Float16* p = A + m * ldA + kk + kb;
  v16h a;
  ((v8h*)&a)[0] = *(const v8h*)p;
  ((v8h*)&a)[1] = *(const v8h*)(p + 8);
  return a;
}
// B (32x16, KxN) from W^T stored N x Kp row-major:
// lane<16 -> K kk..kk+15, lane>=16 -> K kk+16..kk+31, contiguous per lane.
__device__ __forceinline__ v16h load_b_frag(const _Float16* Wt, int ldB, int nc,
                                            int kk, int lane) {
  int n  = nc + (lane & 15);
  int kb = (lane < 16) ? 0 : 16;
  const _Float16* p = Wt + n * ldB + kk + kb;
  v16h b;
  ((v8h*)&b)[0] = *(const v8h*)p;
  ((v8h*)&b)[1] = *(const v8h*)(p + 8);
  return b;
}
__device__ __forceinline__ v8f wmma16(v16h a, v16h b, v8f c) {
  return __builtin_amdgcn_wmma_f32_16x16x32_f16(false, a, false, b, (short)0, c,
                                                false, false);
}

// --- utility kernels --------------------------------------------------------
__global__ void zero_f32_kernel(float* p, int n) {
  int i = blockIdx.x * 256 + threadIdx.x;
  if (i < n) p[i] = 0.0f;
}
__global__ void cast16_kernel(const float* __restrict__ s, _Float16* __restrict__ d, int n) {
  int i = blockIdx.x * 256 + threadIdx.x;
  if (i < n) d[i] = (_Float16)s[i];
}
// transpose + cast: src f32 (K x N) -> dst f16 (N x Kp), zero-pad K..Kp-1
__global__ void wprep_kernel(const float* __restrict__ src, _Float16* __restrict__ dst,
                             int K, int N, int Kp) {
  int i = blockIdx.x * 256 + threadIdx.x;
  if (i >= N * Kp) return;
  int n = i / Kp, k = i - n * Kp;
  dst[i] = (k < K) ? (_Float16)src[k * N + n] : (_Float16)0.0f;
}

// --- generic GEMM: out = [silu](A @ W + b), one 16x16 tile per wave ---------
__global__ void gemm_wmma_kernel(const _Float16* __restrict__ A, int ldA,
                                 const _Float16* __restrict__ Wt, int ldB,
                                 const float* __restrict__ bias,
                                 _Float16* __restrict__ out16, int ldo16,
                                 float* __restrict__ out32, int ldo32,
                                 int Ktot, int doSilu) {
  int lane = threadIdx.x;            // blockDim = 32 (one wave)
  int mr = blockIdx.x * 16;
  int nc = blockIdx.y * 16;
  v8f acc = {};
  for (int kk = 0; kk < Ktot; kk += 32) {
    v16h a  = load_a_frag(A, ldA, mr, kk, lane);
    v16h bf = load_b_frag(Wt, ldB, nc, kk, lane);
    acc = wmma16(a, bf, acc);
  }
  int n  = nc + (lane & 15);
  int mb = (lane >> 4) * 8;
  float bv = bias ? bias[n] : 0.0f;
#pragma unroll
  for (int i = 0; i < 8; ++i) {
    float x = acc[i] + bv;
    if (doSilu) x = silu_f(x);
    int m = mr + mb + i;
    if (out16) out16[m * ldo16 + n] = (_Float16)x;
    if (out32) out32[m * ldo32 + n] = x;
  }
}

// --- pair selection (one block per batch; bitonic top-256 of 2016) ----------
__device__ __forceinline__ void pair_from_index(int t, int& j, int& k) {
  int jj = 0, base = 0;
  while (base + (63 - jj) <= t) { base += 63 - jj; ++jj; }
  j = jj;
  k = jj + 1 + (t - base);
}

__global__ void pair_select_kernel(const float* __restrict__ pos,
                                   const unsigned char* __restrict__ mask,
                                   const int* __restrict__ z,
                                   int* __restrict__ pj, int* __restrict__ pk,
                                   int* __restrict__ pzj, int* __restrict__ pzk,
                                   float* __restrict__ pwArr,
                                   float* __restrict__ pr0j, float* __restrict__ pr0k,
                                   float* __restrict__ prjk, float* __restrict__ pcos,
                                   float* __restrict__ normOut) {
  __shared__ float sx[NN], sy[NN], sz[NN], sr[NN];
  __shared__ int   sval[NN];
  __shared__ float sscore[2048];
  __shared__ int   sidx[2048];
  __shared__ float sred[256];
  int b = blockIdx.x, tid = threadIdx.x;

  if (tid < NN) {
    sx[tid] = pos[(b * NN + tid) * 3 + 0];
    sy[tid] = pos[(b * NN + tid) * 3 + 1];
    sz[tid] = pos[(b * NN + tid) * 3 + 2];
  }
  __syncthreads();
  if (tid < NN) {
    float dx = sx[tid] - sx[0], dy = sy[tid] - sy[0], dz = sz[tid] - sz[0];
    float r = sqrtf(dx * dx + dy * dy + dz * dz);
    sr[tid]   = r;
    sval[tid] = (mask[b * NN + tid] != 0) && (tid != 0) && (r <= CUT);
  }
  __syncthreads();

  for (int t = tid; t < 2048; t += 256) {
    float score = __builtin_inff();
    if (t < NPAIR) {
      int j, k; pair_from_index(t, j, k);
      if (sval[j] && sval[k]) {
        float dx = sx[k] - sx[j], dy = sy[k] - sy[j], dz = sz[k] - sz[j];
        float rjk = sqrtf(dx * dx + dy * dy + dz * dz);
        score = sr[j] + sr[k] + 0.5f * rjk;
      }
    }
    sscore[t] = score;
    sidx[t]   = t;
  }
  __syncthreads();

  // bitonic sort ascending over 2048 entries
  for (int ksz = 2; ksz <= 2048; ksz <<= 1) {
    for (int jsz = ksz >> 1; jsz > 0; jsz >>= 1) {
      for (int t = tid; t < 1024; t += 256) {
        int i   = ((t / jsz) * (jsz << 1)) + (t % jsz);
        int ixj = i + jsz;
        bool up = ((i & ksz) == 0);
        float a = sscore[i], c = sscore[ixj];
        if (up ? (a > c) : (a < c)) {
          sscore[i] = c; sscore[ixj] = a;
          int ti = sidx[i]; sidx[i] = sidx[ixj]; sidx[ixj] = ti;
        }
      }
      __syncthreads();
    }
  }

  // top-256 -> geometry + weights
  {
    int p = tid;
    float s = sscore[p];
    int t = sidx[p];
    int pmask = (s < __builtin_inff()) ? 1 : 0;
    int j = 0, k = 0;
    if (pmask) pair_from_index(t, j, k);
    float vjx = sx[j] - sx[0], vjy = sy[j] - sy[0], vjz = sz[j] - sz[0];
    float vkx = sx[k] - sx[0], vky = sy[k] - sy[0], vkz = sz[k] - sz[0];
    float r0j = sqrtf(vjx * vjx + vjy * vjy + vjz * vjz);
    float r0k = sqrtf(vkx * vkx + vky * vky + vkz * vkz);
    float djx = sx[k] - sx[j], djy = sy[k] - sy[j], djz = sz[k] - sz[j];
    float rjk = sqrtf(djx * djx + djy * djy + djz * djz);
    float ij = 1.0f / fmaxf(r0j, 1e-8f), ik = 1.0f / fmaxf(r0k, 1e-8f);
    float cosang = (vjx * vkx + vjy * vky + vjz * vkz) * ij * ik;
    cosang = fminf(1.0f, fmaxf(-1.0f, cosang));
    float w = cutoff_f(r0j) * cutoff_f(r0k) * cutoff_f(rjk) * (float)pmask;

    int row = b * PP + p;
    pj[row] = j; pk[row] = k;
    pzj[row] = z[b * NN + j]; pzk[row] = z[b * NN + k];
    pwArr[row] = w; pr0j[row] = r0j; pr0k[row] = r0k;
    prjk[row] = rjk; pcos[row] = cosang;
    sred[tid] = w;
  }
  __syncthreads();
  for (int s = 128; s > 0; s >>= 1) {
    if (tid < s) sred[tid] += sred[tid + s];
    __syncthreads();
  }
  if (tid == 0) normOut[b] = fmaxf(sred[0], 1e-8f);
}

// --- build geom_in f16 matrix (4096 x 384): [hj|hk|f0j|f0k|fjk|cos|pad] -----
__global__ void geom_build_kernel(const float* __restrict__ h,
                                  const int* __restrict__ pj, const int* __restrict__ pk,
                                  const float* __restrict__ pr0j, const float* __restrict__ pr0k,
                                  const float* __restrict__ prjk, const float* __restrict__ pcos,
                                  _Float16* __restrict__ Ageom) {
  int row = blockIdx.x;          // B*P rows
  int b   = row >> 8;
  int tid = threadIdx.x;         // 128 threads
  int j = pj[row], k = pk[row];
  _Float16* dst = Ageom + row * KGEOM;
  dst[tid]       = (_Float16)h[(b * NN + j) * HH + tid];
  dst[128 + tid] = (_Float16)h[(b * NN + k) * HH + tid];
  if (tid < 96) {
    float r = (tid < 32) ? pr0j[row] : (tid < 64 ? pr0k[row] : prjk[row]);
    r = fminf(r, CUT);
    int i = tid & 31;
    const float delta = CUT / 31.0f;
    const float gamma = 1.0f / (delta * delta + 1e-12f);
    float d = r - delta * (float)i;
    dst[256 + tid] = (_Float16)__expf(-gamma * d * d);
  }
  if (tid == 96) dst[352] = (_Float16)pcos[row];
  if (tid < 31) dst[353 + tid] = (_Float16)0.0f;   // pad 353..383
}

// --- fused element-MLP (96->128->128->64) + weighted aggregation ------------
// grid: (8 p-chunks, NE/16 e-tiles, B). block: 256 threads = 8 waves.
// Weight fragments are block-invariant -> held in VGPRs for all 32 pairs;
// steady-state inner loop is LDS traffic + v_wmma only.
__global__ void elem_agg_kernel(const _Float16* __restrict__ zemb16,   // 91 x 32
                                const _Float16* __restrict__ efeat16,  // 128 x 32
                                const _Float16* __restrict__ pw0t,     // 128 x 96
                                const float* __restrict__ pb0,
                                const _Float16* __restrict__ pw1t,     // 128 x 128
                                const float* __restrict__ pb1,
                                const _Float16* __restrict__ pw2t,     // 64 x 128
                                const float* __restrict__ pb2,
                                const int* __restrict__ pzj, const int* __restrict__ pzk,
                                const float* __restrict__ pwArr,
                                const float* __restrict__ ggeom,       // 4096 x 64 f32
                                float* __restrict__ agg) {             // B x 128 x 64
  __shared__ __attribute__((aligned(16))) _Float16 sIn[16 * 96];
  __shared__ __attribute__((aligned(16))) _Float16 sX1[16 * 128];
  __shared__ __attribute__((aligned(16))) _Float16 sX2[16 * 128];
  __shared__ float sAcc[16 * 64];

  int tid  = threadIdx.x;
  int lane = tid & 31;
  int wv   = tid >> 5;
  int pc = blockIdx.x, et = blockIdx.y, b = blockIdx.z;
  int nc = wv * 16;
  int nl = nc + (lane & 15);
  int mb = (lane >> 4) * 8;

  // hoist block-invariant weight fragments + biases into registers
  v16h w0f[3], w1f[4], w2f[4];
#pragma unroll
  for (int ki = 0; ki < 3; ++ki) w0f[ki] = load_b_frag(pw0t, 96, nc, ki * 32, lane);
#pragma unroll
  for (int ki = 0; ki < 4; ++ki) w1f[ki] = load_b_frag(pw1t, 128, nc, ki * 32, lane);
  if (wv < 4) {
#pragma unroll
    for (int ki = 0; ki < 4; ++ki) w2f[ki] = load_b_frag(pw2t, 128, nc, ki * 32, lane);
  } else {
#pragma unroll
    for (int ki = 0; ki < 4; ++ki) w2f[ki] = v16h{};
  }
  float b0 = pb0[nl];
  float b1 = pb1[nl];
  float b2 = (wv < 4) ? pb2[nl] : 0.0f;

  for (int i = tid; i < 16 * 64; i += 256) sAcc[i] = 0.0f;
  for (int i = tid; i < 16 * 32; i += 256) {     // e_feat cols 64..95, fixed
    int rr = i >> 5, cc = i & 31;
    sIn[rr * 96 + 64 + cc] = efeat16[(et * 16 + rr) * 32 + cc];
  }
  __syncthreads();

  for (int pp = 0; pp < 32; ++pp) {
    int prow = b * PP + pc * 32 + pp;
    float wpair = pwArr[prow];
    if (wpair != 0.0f) {                         // block-uniform branch
      int zj = pzj[prow], zk = pzk[prow];
      for (int i = tid; i < 16 * 64; i += 256) { // zj|zk embeddings, cols 0..63
        int rr = i >> 6, cc = i & 63;
        _Float16 v = (cc < 32) ? zemb16[zj * 32 + cc] : zemb16[zk * 32 + (cc - 32)];
        sIn[rr * 96 + cc] = v;
      }
      __syncthreads();
      // layer 1: 96 -> 128
      {
        v8f acc = {};
#pragma unroll
        for (int ki = 0; ki < 3; ++ki)
          acc = wmma16(load_a_frag(sIn, 96, 0, ki * 32, lane), w0f[ki], acc);
#pragma unroll
        for (int i = 0; i < 8; ++i)
          sX1[(mb + i) * 128 + nl] = (_Float16)silu_f(acc[i] + b0);
      }
      __syncthreads();
      // layer 2: 128 -> 128
      {
        v8f acc = {};
#pragma unroll
        for (int ki = 0; ki < 4; ++ki)
          acc = wmma16(load_a_frag(sX1, 128, 0, ki * 32, lane), w1f[ki], acc);
#pragma unroll
        for (int i = 0; i < 8; ++i)
          sX2[(mb + i) * 128 + nl] = (_Float16)silu_f(acc[i] + b1);
      }
      __syncthreads();
      // layer 3: 128 -> 64, scale by w * g_geom, accumulate
      if (wv < 4) {
        v8f acc = {};
#pragma unroll
        for (int ki = 0; ki < 4; ++ki)
          acc = wmma16(load_a_frag(sX2, 128, 0, ki * 32, lane), w2f[ki], acc);
        float scale = wpair * ggeom[prow * 64 + nl];
#pragma unroll
        for (int i = 0; i < 8; ++i)
          sAcc[(mb + i) * 64 + nl] += scale * (acc[i] + b2);
      }
      __syncthreads();
    }
  }
  __syncthreads();
  for (int i = tid; i < 16 * 64; i += 256) {
    int rr = i >> 6, cc = i & 63;
    atomicAdd(&agg[((b * NE) + et * 16 + rr) * 64 + cc], sAcc[rr * 64 + cc]);
  }
}

// --- agg / norm -> f16 for output MLP ---------------------------------------
__global__ void norm_cast_kernel(const float* __restrict__ agg,
                                 const float* __restrict__ normArr,
                                 _Float16* __restrict__ out) {
  int i = blockIdx.x * 256 + threadIdx.x;        // over B*NE*64
  if (i < BB * NE * 64) {
    int b = i / (NE * 64);
    out[i] = (_Float16)(agg[i] / normArr[b]);
  }
}

// ---------------------------------------------------------------------------
static size_t carve_off(size_t& off, size_t bytes) {
  size_t r = off;
  off += (bytes + 255) & ~(size_t)255;
  return r;
}

extern "C" void kernel_launch(void* const* d_in, const int* in_sizes, int n_in,
                              void* d_out, int out_size, void* d_ws, size_t ws_size,
                              hipStream_t stream) {
  const float* h      = (const float*)d_in[0];
  const int*   z      = (const int*)d_in[1];
  const float* pos    = (const float*)d_in[2];
  const unsigned char* mask = (const unsigned char*)d_in[3];
  const float* e_feat = (const float*)d_in[4];
  const float* z_emb  = (const float*)d_in[5];
  const float* pw0 = (const float*)d_in[6];  const float* pb0 = (const float*)d_in[7];
  const float* pw1 = (const float*)d_in[8];  const float* pb1 = (const float*)d_in[9];
  const float* pw2 = (const float*)d_in[10]; const float* pb2 = (const float*)d_in[11];
  const float* gw0 = (const float*)d_in[12]; const float* gb0 = (const float*)d_in[13];
  const float* gw1 = (const float*)d_in[14]; const float* gb1 = (const float*)d_in[15];
  const float* gw2 = (const float*)d_in[16]; const float* gb2 = (const float*)d_in[17];
  const float* ow0 = (const float*)d_in[18]; const float* ob0 = (const float*)d_in[19];
  const float* ow1 = (const float*)d_in[20]; const float* ob1 = (const float*)d_in[21];
  float* out = (float*)d_out;                // (B, nE, 128)

  char* ws = (char*)d_ws;
  size_t off = 0;
  const int NR = BB * PP;                    // 4096 pair rows
  int*   pj    = (int*)(ws + carve_off(off, NR * 4));
  int*   pk    = (int*)(ws + carve_off(off, NR * 4));
  int*   pzj   = (int*)(ws + carve_off(off, NR * 4));
  int*   pzk   = (int*)(ws + carve_off(off, NR * 4));
  float* pwA   = (float*)(ws + carve_off(off, NR * 4));
  float* pr0j  = (float*)(ws + carve_off(off, NR * 4));
  float* pr0k  = (float*)(ws + carve_off(off, NR * 4));
  float* prjk  = (float*)(ws + carve_off(off, NR * 4));
  float* pcos  = (float*)(ws + carve_off(off, NR * 4));
  float* normA = (float*)(ws + carve_off(off, BB * 4));

  _Float16* gw0t = (_Float16*)(ws + carve_off(off, 256 * 384 * 2));
  _Float16* gw1t = (_Float16*)(ws + carve_off(off, 256 * 256 * 2));
  _Float16* gw2t = (_Float16*)(ws + carve_off(off, 64 * 256 * 2));
  _Float16* pw0t = (_Float16*)(ws + carve_off(off, 128 * 96 * 2));
  _Float16* pw1t = (_Float16*)(ws + carve_off(off, 128 * 128 * 2));
  _Float16* pw2t = (_Float16*)(ws + carve_off(off, 64 * 128 * 2));
  _Float16* ow0t = (_Float16*)(ws + carve_off(off, 256 * 64 * 2));
  _Float16* ow1t = (_Float16*)(ws + carve_off(off, 128 * 256 * 2));
  _Float16* zemb16  = (_Float16*)(ws + carve_off(off, 91 * 32 * 2));
  _Float16* efeat16 = (_Float16*)(ws + carve_off(off, 128 * 32 * 2));

  _Float16* Ageom = (_Float16*)(ws + carve_off(off, (size_t)NR * KGEOM * 2));
  _Float16* G1    = (_Float16*)(ws + carve_off(off, (size_t)NR * 256 * 2));
  _Float16* G2    = (_Float16*)(ws + carve_off(off, (size_t)NR * 256 * 2));
  float*    ggeom = (float*)(ws + carve_off(off, (size_t)NR * 64 * 4));
  float*    agg   = (float*)(ws + carve_off(off, (size_t)BB * NE * 64 * 4));
  _Float16* aggN  = (_Float16*)(ws + carve_off(off, (size_t)BB * NE * 64 * 2));
  _Float16* Xo    = (_Float16*)(ws + carve_off(off, (size_t)BB * NE * 256 * 2));
  (void)ws_size; (void)n_in; (void)in_sizes; (void)out_size;

  // 1) weight prep (transpose + f32->f16, K zero-padded)
  auto wprep = [&](const float* s, _Float16* d, int K, int Nn, int Kp) {
    int tot = Nn * Kp;
    wprep_kernel<<<(tot + 255) / 256, 256, 0, stream>>>(s, d, K, Nn, Kp);
  };
  wprep(gw0, gw0t, 353, 256, 384);
  wprep(gw1, gw1t, 256, 256, 256);
  wprep(gw2, gw2t, 256, 64, 256);
  wprep(pw0, pw0t, 96, 128, 96);
  wprep(pw1, pw1t, 128, 128, 128);
  wprep(pw2, pw2t, 128, 64, 128);
  wprep(ow0, ow0t, 64, 256, 64);
  wprep(ow1, ow1t, 256, 128, 256);
  cast16_kernel<<<(91 * 32 + 255) / 256, 256, 0, stream>>>(z_emb, zemb16, 91 * 32);
  cast16_kernel<<<(128 * 32 + 255) / 256, 256, 0, stream>>>(e_feat, efeat16, 128 * 32);

  // 2) pair selection + geometry scalars + per-batch norm
  pair_select_kernel<<<BB, 256, 0, stream>>>(pos, mask, z, pj, pk, pzj, pzk,
                                             pwA, pr0j, pr0k, prjk, pcos, normA);

  // 3) geometry MLP: build A (4096x384) then 384->256->256->64 via WMMA
  geom_build_kernel<<<NR, 128, 0, stream>>>(h, pj, pk, pr0j, pr0k, prjk, pcos, Ageom);
  gemm_wmma_kernel<<<dim3(NR / 16, 256 / 16), 32, 0, stream>>>(
      Ageom, KGEOM, gw0t, 384, gb0, G1, 256, nullptr, 0, 384, 1);
  gemm_wmma_kernel<<<dim3(NR / 16, 256 / 16), 32, 0, stream>>>(
      G1, 256, gw1t, 256, gb1, G2, 256, nullptr, 0, 256, 1);
  gemm_wmma_kernel<<<dim3(NR / 16, 64 / 16), 32, 0, stream>>>(
      G2, 256, gw2t, 256, gb2, nullptr, 0, ggeom, 64, 256, 0);

  // 4) fused element MLP + weighted aggregation over pairs
  zero_f32_kernel<<<(BB * NE * 64 + 255) / 256, 256, 0, stream>>>(agg, BB * NE * 64);
  elem_agg_kernel<<<dim3(PP / 32, NE / 16, BB), 256, 0, stream>>>(
      zemb16, efeat16, pw0t, pb0, pw1t, pb1, pw2t, pb2,
      pzj, pzk, pwA, ggeom, agg);

  // 5) normalize + output MLP: 64 -> 256 (silu) -> 128
  norm_cast_kernel<<<(BB * NE * 64 + 255) / 256, 256, 0, stream>>>(agg, normA, aggN);
  gemm_wmma_kernel<<<dim3(BB * NE / 16, 256 / 16), 32, 0, stream>>>(
      aggN, 64, ow0t, 64, ob0, Xo, 256, nullptr, 0, 64, 1);
  gemm_wmma_kernel<<<dim3(BB * NE / 16, 128 / 16), 32, 0, stream>>>(
      Xo, 256, ow1t, 256, ob1, nullptr, 0, out, 128, 256, 0);
}
